// Graphnetwork_phonon_77824807403565
// MI455X (gfx1250) — matrix operand order repaired and checked by reading, not compiled, not run on hardware
//
#include <hip/hip_runtime.h>
#include <hip/hip_bf16.h>
#include <math.h>

typedef __attribute__((ext_vector_type(16))) _Float16 v16h;
typedef __attribute__((ext_vector_type(8)))  _Float16 v8h;
typedef __attribute__((ext_vector_type(8)))  float    v8f;

#define GN_N   50000
#define GN_E   400000
#define GN_H   64
#define GN_L   3
#define GN_G   256
#define GN_NA  118
#define GN_NE  51

static inline int gn_cdiv(long long a, long long b) { return (int)((a + b - 1) / b); }

// ---------------------------------------------------------------------------
// Weight repack: row-major f32 [K,N] -> per-lane WMMA B-fragment layout f16.
// Packed index p = ((kb*NT + nt)*32 + lane)*16 + i, where
//   k = kb*32 + (i>>3)*16 + (lane>>4)*8 + (i&7) , n = nt*16 + (lane&15).
// Zero-pads K up to Kpad (multiple of 32).
// ---------------------------------------------------------------------------
__global__ void gn_repack_w(const float* __restrict__ W, _Float16* __restrict__ Wp,
                            int K, int Kpad, int N) {
    int idx = blockIdx.x * blockDim.x + threadIdx.x;
    int total = Kpad * N;
    if (idx >= total) return;
    int i    = idx & 15;
    int lane = (idx >> 4) & 31;
    int rest = idx >> 9;
    int NT   = N >> 4;
    int nt   = rest % NT;
    int kb   = rest / NT;
    int k = kb * 32 + (i >> 3) * 16 + (lane >> 4) * 8 + (i & 7);
    int n = nt * 16 + (lane & 15);
    float v = (k < K) ? W[(size_t)k * N + n] : 0.0f;
    Wp[idx] = (_Float16)v;
}

// ---------------------------------------------------------------------------
// Core: one wave computes a 16-row strip x (NT*16) cols, K = KB*32, f16 inputs.
// A fragment per lane: lanes 0-15 hold K {0..7,16..23}, lanes 16-31 {8..15,24..31}.
// ---------------------------------------------------------------------------
template<int NT, int KB>
__device__ __forceinline__ void gn_wmma_strip(const _Float16* __restrict__ A,
                                              const _Float16* __restrict__ Wp,
                                              int tile, v8f acc[NT]) {
    const int lane = threadIdx.x & 31;
    const int K = KB * 32;
    const _Float16* arow = A + (size_t)(tile * 16 + (lane & 15)) * K + ((lane & 16) ? 8 : 0);
    const _Float16* wl   = Wp + (size_t)lane * 16;
#pragma unroll
    for (int kb = 0; kb < KB; ++kb) {
        union { v16h v; v8h h[2]; } af;
        af.h[0] = *(const v8h*)(arow + kb * 32);
        af.h[1] = *(const v8h*)(arow + kb * 32 + 16);
#pragma unroll
        for (int nt = 0; nt < NT; ++nt) {
            v16h bf = *(const v16h*)(wl + (size_t)(kb * NT + nt) * 512);
            acc[nt] = __builtin_amdgcn_wmma_f32_16x16x32_f16(
                false, af.v, false, bf, (short)0, acc[nt], false, false);
        }
    }
}

// Shared epilogue: +bias -> LayerNorm(row) -> PReLU(scalar) -> f16 store.
// Rows 0..7 live in lanes 0..15 (acc elem j), rows 8..15 in lanes 16..31;
// row stats via half-wave shfl_xor (masks 8,4,2,1 stay within the half).
template<int NT>
__device__ __forceinline__ void gn_ln_prelu_store(v8f acc[NT], const float* __restrict__ bias,
                                                  const float* __restrict__ gamma,
                                                  const float* __restrict__ beta,
                                                  float p, _Float16* __restrict__ out, int tile) {
    const int lane = threadIdx.x & 31;
    const int nlo  = lane & 15;
    const int Ncol = NT * 16;
#pragma unroll
    for (int nt = 0; nt < NT; ++nt) {
        float b = bias[nt * 16 + nlo];
#pragma unroll
        for (int j = 0; j < 8; ++j) acc[nt][j] += b;
    }
    float mean[8], rstd[8];
#pragma unroll
    for (int j = 0; j < 8; ++j) {
        float s = 0.0f, s2 = 0.0f;
#pragma unroll
        for (int nt = 0; nt < NT; ++nt) { float v = acc[nt][j]; s += v; s2 += v * v; }
        s  += __shfl_xor(s, 8, 32);  s2 += __shfl_xor(s2, 8, 32);
        s  += __shfl_xor(s, 4, 32);  s2 += __shfl_xor(s2, 4, 32);
        s  += __shfl_xor(s, 2, 32);  s2 += __shfl_xor(s2, 2, 32);
        s  += __shfl_xor(s, 1, 32);  s2 += __shfl_xor(s2, 1, 32);
        float m = s * (1.0f / Ncol);
        mean[j] = m;
        rstd[j] = rsqrtf(s2 * (1.0f / Ncol) - m * m + 1e-5f);
    }
    const int mbase = tile * 16 + ((lane & 16) ? 8 : 0);
#pragma unroll
    for (int nt = 0; nt < NT; ++nt) {
        int c = nt * 16 + nlo;
        float g = gamma[c], be = beta[c];
#pragma unroll
        for (int j = 0; j < 8; ++j) {
            float v = (acc[nt][j] - mean[j]) * rstd[j] * g + be;
            v = (v >= 0.0f) ? v : p * v;
            out[(size_t)(mbase + j) * Ncol + c] = (_Float16)v;
        }
    }
}

// GEMM + bias -> f32 out
template<int NT, int KB>
__global__ void gn_gemm_bias_f32(const _Float16* __restrict__ A, const _Float16* __restrict__ Wp,
                                 const float* __restrict__ bias, float* __restrict__ out,
                                 int Mtiles) {
    int tile = blockIdx.x * (blockDim.x >> 5) + (threadIdx.x >> 5);
    if (tile >= Mtiles) return;
    v8f acc[NT] = {};
    gn_wmma_strip<NT, KB>(A, Wp, tile, acc);
    const int lane = threadIdx.x & 31;
    const int nlo  = lane & 15;
    const int mbase = tile * 16 + ((lane & 16) ? 8 : 0);
    const int Ncol = NT * 16;
#pragma unroll
    for (int nt = 0; nt < NT; ++nt) {
        int c = nt * 16 + nlo;
        float b = bias[c];
#pragma unroll
        for (int j = 0; j < 8; ++j)
            out[(size_t)(mbase + j) * Ncol + c] = acc[nt][j] + b;
    }
}

// GEMM + bias -> f32 AND f16 outputs in one pass (encoder second layers)
template<int NT, int KB>
__global__ void gn_gemm_bias_dual(const _Float16* __restrict__ A, const _Float16* __restrict__ Wp,
                                  const float* __restrict__ bias, float* __restrict__ out32,
                                  _Float16* __restrict__ out16, int Mtiles) {
    int tile = blockIdx.x * (blockDim.x >> 5) + (threadIdx.x >> 5);
    if (tile >= Mtiles) return;
    v8f acc[NT] = {};
    gn_wmma_strip<NT, KB>(A, Wp, tile, acc);
    const int lane = threadIdx.x & 31;
    const int nlo  = lane & 15;
    const int mbase = tile * 16 + ((lane & 16) ? 8 : 0);
    const int Ncol = NT * 16;
#pragma unroll
    for (int nt = 0; nt < NT; ++nt) {
        int c = nt * 16 + nlo;
        float b = bias[c];
#pragma unroll
        for (int j = 0; j < 8; ++j) {
            float v = acc[nt][j] + b;
            size_t idx = (size_t)(mbase + j) * Ncol + c;
            out32[idx] = v;
            out16[idx] = (_Float16)v;
        }
    }
}

// GEMM + bias + LeakyReLU(0.01) -> f32 out
template<int NT, int KB>
__global__ void gn_gemm_bias_leaky_f32(const _Float16* __restrict__ A, const _Float16* __restrict__ Wp,
                                       const float* __restrict__ bias, float* __restrict__ out,
                                       int Mtiles) {
    int tile = blockIdx.x * (blockDim.x >> 5) + (threadIdx.x >> 5);
    if (tile >= Mtiles) return;
    v8f acc[NT] = {};
    gn_wmma_strip<NT, KB>(A, Wp, tile, acc);
    const int lane = threadIdx.x & 31;
    const int nlo  = lane & 15;
    const int mbase = tile * 16 + ((lane & 16) ? 8 : 0);
    const int Ncol = NT * 16;
#pragma unroll
    for (int nt = 0; nt < NT; ++nt) {
        int c = nt * 16 + nlo;
        float b = bias[c];
#pragma unroll
        for (int j = 0; j < 8; ++j) {
            float v = acc[nt][j] + b;
            out[(size_t)(mbase + j) * Ncol + c] = (v >= 0.0f) ? v : 0.01f * v;
        }
    }
}

// GEMM + bias + PReLU(scalar) -> f16 out
template<int NT, int KB>
__global__ void gn_gemm_bias_prelu_f16(const _Float16* __restrict__ A, const _Float16* __restrict__ Wp,
                                       const float* __restrict__ bias, const float* __restrict__ pp,
                                       _Float16* __restrict__ out, int Mtiles) {
    int tile = blockIdx.x * (blockDim.x >> 5) + (threadIdx.x >> 5);
    if (tile >= Mtiles) return;
    v8f acc[NT] = {};
    gn_wmma_strip<NT, KB>(A, Wp, tile, acc);
    const int lane = threadIdx.x & 31;
    const int nlo  = lane & 15;
    const int mbase = tile * 16 + ((lane & 16) ? 8 : 0);
    const int Ncol = NT * 16;
    const float p = pp[0];
#pragma unroll
    for (int nt = 0; nt < NT; ++nt) {
        int c = nt * 16 + nlo;
        float b = bias[c];
#pragma unroll
        for (int j = 0; j < 8; ++j) {
            float v = acc[nt][j] + b;
            v = (v >= 0.0f) ? v : p * v;
            out[(size_t)(mbase + j) * Ncol + c] = (_Float16)v;
        }
    }
}

// ---------------------------------------------------------------------------
// FUSED edge MLP layer-1: A row e = concat(xh16[row[e]], xh16[col[e]], ea16[e]),
// gathered directly in A-fragment address generation (no e_in materialization).
// K = 192 (KB=6: kb 0-1 -> xh[row], 2-3 -> xh[col], 4-5 -> ea), N = 128 (NT=8).
// ---------------------------------------------------------------------------
__global__ void gn_edge_gemm1_ln_prelu(const _Float16* __restrict__ xh16,
                                       const _Float16* __restrict__ ea16,
                                       const int* __restrict__ row, const int* __restrict__ col,
                                       const _Float16* __restrict__ Wp,
                                       const float* __restrict__ bias,
                                       const float* __restrict__ gamma, const float* __restrict__ beta,
                                       const float* __restrict__ pp,
                                       _Float16* __restrict__ out, int Mtiles) {
    constexpr int NT = 8, KB = 6;
    int tile = blockIdx.x * (blockDim.x >> 5) + (threadIdx.x >> 5);
    if (tile >= Mtiles) return;
    const int lane = threadIdx.x & 31;
    const int e = tile * 16 + (lane & 15);
    const int koff = (lane & 16) ? 8 : 0;
    const _Float16* segs[3] = {
        xh16 + (size_t)row[e] * 64 + koff,
        xh16 + (size_t)col[e] * 64 + koff,
        ea16 + (size_t)e      * 64 + koff
    };
    const _Float16* wl = Wp + (size_t)lane * 16;
    v8f acc[NT] = {};
#pragma unroll
    for (int kb = 0; kb < KB; ++kb) {
        const _Float16* p = segs[kb >> 1] + (kb & 1) * 32;
        union { v16h v; v8h h[2]; } af;
        af.h[0] = *(const v8h*)(p);
        af.h[1] = *(const v8h*)(p + 16);
#pragma unroll
        for (int nt = 0; nt < NT; ++nt) {
            v16h bf = *(const v16h*)(wl + (size_t)(kb * NT + nt) * 512);
            acc[nt] = __builtin_amdgcn_wmma_f32_16x16x32_f16(
                false, af.v, false, bf, (short)0, acc[nt], false, false);
        }
    }
    gn_ln_prelu_store<NT>(acc, bias, gamma, beta, pp[0], out, tile);
}

// ---------------------------------------------------------------------------
// FUSED edge MLP layer-2: e_out = h1 @ W2 + b2 held in registers; directly
//   (a) atomicAdd into agg[col[e]] (segment-sum for scatter-mean)
//   (b) ea += e_out residual, with f16 shadow copy.
// No e_out materialization. K = 128 (KB=4), N = 64 (NT=4).
// ---------------------------------------------------------------------------
__global__ void gn_edge_gemm2_fused(const _Float16* __restrict__ A, const _Float16* __restrict__ Wp,
                                    const float* __restrict__ bias, const int* __restrict__ col,
                                    float* __restrict__ agg, float* __restrict__ ea,
                                    _Float16* __restrict__ ea16, int Mtiles) {
    constexpr int NT = 4, KB = 4;
    int tile = blockIdx.x * (blockDim.x >> 5) + (threadIdx.x >> 5);
    if (tile >= Mtiles) return;
    v8f acc[NT] = {};
    gn_wmma_strip<NT, KB>(A, Wp, tile, acc);
    const int lane = threadIdx.x & 31;
    const int nlo  = lane & 15;
    const int mbase = tile * 16 + ((lane & 16) ? 8 : 0);
#pragma unroll
    for (int j = 0; j < 8; ++j) {
        int e  = mbase + j;
        int ce = col[e];
#pragma unroll
        for (int nt = 0; nt < NT; ++nt) {
            int c = nt * 16 + nlo;
            float v = acc[nt][j] + bias[c];
            atomicAdd(&agg[(size_t)ce * 64 + c], v);
            size_t idx = (size_t)e * 64 + c;
            float nv = ea[idx] + v;
            ea[idx]   = nv;
            ea16[idx] = (_Float16)nv;
        }
    }
}

// ---------------------------------------------------------------------------
// FUSED node MLP layer-1: A row n = concat(xh16[n], agg[n]/max(cnt[n],1)),
// f32 agg converted + scaled in-register (scatter-mean is linear -> exact).
// K = 128 (KB=4: kb 0-1 -> xh16, 2-3 -> agg), N = 128 (NT=8).
// ---------------------------------------------------------------------------
__global__ void gn_node_gemm1_ln_prelu(const _Float16* __restrict__ xh16,
                                       const float* __restrict__ agg, const float* __restrict__ cnt,
                                       const _Float16* __restrict__ Wp,
                                       const float* __restrict__ bias,
                                       const float* __restrict__ gamma, const float* __restrict__ beta,
                                       const float* __restrict__ pp,
                                       _Float16* __restrict__ out, int Mtiles) {
    constexpr int NT = 8, KB = 4;
    int tile = blockIdx.x * (blockDim.x >> 5) + (threadIdx.x >> 5);
    if (tile >= Mtiles) return;
    const int lane = threadIdx.x & 31;
    const int n = tile * 16 + (lane & 15);
    const int koff = (lane & 16) ? 8 : 0;
    const float ic = 1.0f / fmaxf(cnt[n], 1.0f);
    const _Float16* xrow = xh16 + (size_t)n * 64 + koff;
    const float*    arow = agg  + (size_t)n * 64 + koff;
    const _Float16* wl = Wp + (size_t)lane * 16;
    v8f acc[NT] = {};
#pragma unroll
    for (int kb = 0; kb < KB; ++kb) {
        union { v16h v; v8h h[2]; } af;
        if (kb < 2) {
            const _Float16* p = xrow + kb * 32;
            af.h[0] = *(const v8h*)(p);
            af.h[1] = *(const v8h*)(p + 16);
        } else {
            const float* q = arow + (kb - 2) * 32;
            v8f f0 = *(const v8f*)(q);
            v8f f1 = *(const v8f*)(q + 16);
#pragma unroll
            for (int i = 0; i < 8; ++i) {
                af.h[0][i] = (_Float16)(f0[i] * ic);
                af.h[1][i] = (_Float16)(f1[i] * ic);
            }
        }
#pragma unroll
        for (int nt = 0; nt < NT; ++nt) {
            v16h bf = *(const v16h*)(wl + (size_t)(kb * NT + nt) * 512);
            acc[nt] = __builtin_amdgcn_wmma_f32_16x16x32_f16(
                false, af.v, false, bf, (short)0, acc[nt], false, false);
        }
    }
    gn_ln_prelu_store<NT>(acc, bias, gamma, beta, pp[0], out, tile);
}

// ---------------------------------------------------------------------------
// FUSED node MLP layer-2: x_out held in registers; xh += x_out with f16 shadow.
// K = 128 (KB=4), N = 64 (NT=4).
// ---------------------------------------------------------------------------
__global__ void gn_node_gemm2_fused(const _Float16* __restrict__ A, const _Float16* __restrict__ Wp,
                                    const float* __restrict__ bias,
                                    float* __restrict__ xh, _Float16* __restrict__ xh16,
                                    int Mtiles) {
    constexpr int NT = 4, KB = 4;
    int tile = blockIdx.x * (blockDim.x >> 5) + (threadIdx.x >> 5);
    if (tile >= Mtiles) return;
    v8f acc[NT] = {};
    gn_wmma_strip<NT, KB>(A, Wp, tile, acc);
    const int lane = threadIdx.x & 31;
    const int nlo  = lane & 15;
    const int mbase = tile * 16 + ((lane & 16) ? 8 : 0);
#pragma unroll
    for (int nt = 0; nt < NT; ++nt) {
        int c = nt * 16 + nlo;
        float b = bias[c];
#pragma unroll
        for (int j = 0; j < 8; ++j) {
            size_t idx = (size_t)(mbase + j) * 64 + c;
            float nv = xh[idx] + acc[nt][j] + b;
            xh[idx]   = nv;
            xh16[idx] = (_Float16)nv;
        }
    }
}

// ---------------------------------------------------------------------------
// Elementwise / scatter helpers
// ---------------------------------------------------------------------------
__global__ void gn_f32_to_f16(const float* __restrict__ in, _Float16* __restrict__ out, size_t n) {
    size_t i = (size_t)blockIdx.x * blockDim.x + threadIdx.x;
    if (i < n) out[i] = (_Float16)in[i];
}

__global__ void gn_pad_x(const float* __restrict__ x, _Float16* __restrict__ xp) {
    size_t i = (size_t)blockIdx.x * blockDim.x + threadIdx.x;
    if (i >= (size_t)GN_N * 128) return;
    int n = (int)(i >> 7), k = (int)(i & 127);
    xp[i] = (k < GN_NA) ? (_Float16)x[(size_t)n * GN_NA + k] : (_Float16)0.0f;
}

__global__ void gn_edge_feat(const float* __restrict__ ev, float* __restrict__ ef) {
    int e = blockIdx.x * blockDim.x + threadIdx.x;
    if (e >= GN_E) return;
    float x = ev[(size_t)e * 3 + 0], y = ev[(size_t)e * 3 + 1], z = ev[(size_t)e * 3 + 2];
    float r = sqrtf(x * x + y * y + z * z);
    float inv = 1.0f / r;
    float u = 0.5f * r - 2.0f;
    float cut = (u > 0.0f) ? 0.0f
              : ((u < -1.0f) ? 1.0f : 0.5f * (1.0f - cosf(3.14159265358979323846f * u)));
    const float s3 = 1.7320508075688772f;
    ef[(size_t)e * 4 + 0] = cut;
    ef[(size_t)e * 4 + 1] = cut * s3 * y * inv;
    ef[(size_t)e * 4 + 2] = cut * s3 * z * inv;
    ef[(size_t)e * 4 + 3] = cut * s3 * x * inv;
}

// edge-encoder layer1 (K=4, too small for WMMA): prelu(ef @ w1 + b1) -> f16
__global__ void gn_ee1(const float* __restrict__ ef, const float* __restrict__ w1,
                       const float* __restrict__ b1, const float* __restrict__ pp,
                       _Float16* __restrict__ out) {
    size_t i = (size_t)blockIdx.x * blockDim.x + threadIdx.x;
    if (i >= (size_t)GN_E * 64) return;
    int e = (int)(i >> 6), j = (int)(i & 63);
    float s = b1[j];
#pragma unroll
    for (int k = 0; k < 4; ++k) s += ef[(size_t)e * 4 + k] * w1[k * 64 + j];
    float p = pp[0];
    s = (s >= 0.0f) ? s : p * s;
    out[i] = (_Float16)s;
}

__global__ void gn_count(const int* __restrict__ col, float* __restrict__ cnt) {
    int e = blockIdx.x * blockDim.x + threadIdx.x;
    if (e < GN_E) atomicAdd(&cnt[col[e]], 1.0f);
}

__global__ void gn_scatter_batch(const float* __restrict__ xh, const int* __restrict__ batch,
                                 float* __restrict__ gsum) {
    size_t i = (size_t)blockIdx.x * blockDim.x + threadIdx.x;
    if (i >= (size_t)GN_N * 64) return;
    int n = (int)(i >> 6), j = (int)(i & 63);
    atomicAdd(&gsum[(size_t)batch[n] * 64 + j], xh[i]);
}

__global__ void gn_build_dos_in(const _Float16* __restrict__ emb16, const _Float16* __restrict__ graph16,
                                _Float16* __restrict__ dos_in) {
    size_t i = (size_t)blockIdx.x * blockDim.x + threadIdx.x;
    if (i >= (size_t)GN_NE * GN_G * 64) return;
    int m = (int)(i >> 6), j = (int)(i & 63);
    int ne = m >> 8, g = m & 255;   // G == 256
    dos_in[(size_t)m * 128 + j]      = emb16[ne * 64 + j];
    dos_in[(size_t)m * 128 + 64 + j] = graph16[g * 64 + j];
}

__global__ void gn_final(const float* __restrict__ h, const float* __restrict__ w2,
                         const float* __restrict__ b2, float* __restrict__ out) {
    int m = blockIdx.x * blockDim.x + threadIdx.x;
    if (m >= GN_NE * GN_G) return;
    int ne = m >> 8, g = m & 255;
    float s = b2[0];
#pragma unroll
    for (int j = 0; j < 64; ++j) s += h[(size_t)m * 64 + j] * w2[j];
    out[(size_t)g * GN_NE + ne] = s;
}

// ---------------------------------------------------------------------------
extern "C" void kernel_launch(void* const* d_in, const int* in_sizes, int n_in,
                              void* d_out, int out_size, void* d_ws, size_t ws_size,
                              hipStream_t stream) {
    const float* x        = (const float*)d_in[0];
    const float* edge_vec = (const float*)d_in[1];
    const int*   ei       = (const int*)  d_in[2];
    const int*   batch    = (const int*)  d_in[3];
    const float* emb      = (const float*)d_in[4];
    const float* ne_w1 = (const float*)d_in[5];  const float* ne_b1 = (const float*)d_in[6];
    const float* ne_p1 = (const float*)d_in[7];
    const float* ne_w2 = (const float*)d_in[8];  const float* ne_b2 = (const float*)d_in[9];
    const float* ee_w1 = (const float*)d_in[10]; const float* ee_b1 = (const float*)d_in[11];
    const float* ee_p1 = (const float*)d_in[12];
    const float* ee_w2 = (const float*)d_in[13]; const float* ee_b2 = (const float*)d_in[14];
    const float* em_w1 = (const float*)d_in[15]; const float* em_b1 = (const float*)d_in[16];
    const float* em_g  = (const float*)d_in[17]; const float* em_be = (const float*)d_in[18];
    const float* em_p  = (const float*)d_in[19];
    const float* em_w2 = (const float*)d_in[20]; const float* em_b2 = (const float*)d_in[21];
    const float* nm_w1 = (const float*)d_in[22]; const float* nm_b1 = (const float*)d_in[23];
    const float* nm_g  = (const float*)d_in[24]; const float* nm_be = (const float*)d_in[25];
    const float* nm_p  = (const float*)d_in[26];
    const float* nm_w2 = (const float*)d_in[27]; const float* nm_b2 = (const float*)d_in[28];
    const float* dec_w = (const float*)d_in[29]; const float* dec_b = (const float*)d_in[30];
    const float* out_w1 = (const float*)d_in[31]; const float* out_b1 = (const float*)d_in[32];
    const float* out_w2 = (const float*)d_in[33]; const float* out_b2 = (const float*)d_in[34];
    float* out = (float*)d_out;

    const int* rowp = ei;
    const int* colp = ei + GN_E;

    // ---- workspace carving (256B aligned) ----
    char* ws = (char*)d_ws;
    size_t off = 0;
    auto alloc = [&](size_t bytes) -> void* {
        void* p = ws + off;
        off += (bytes + 255) & ~(size_t)255;
        return p;
    };
    _Float16* Wp_ne1 = (_Float16*)alloc(128 * 64 * 2);
    _Float16* Wp_ne2 = (_Float16*)alloc(64 * 64 * 2);
    _Float16* Wp_ee2 = (_Float16*)alloc(64 * 64 * 2);
    _Float16* Wp_em1 = (_Float16*)alloc((size_t)GN_L * 192 * 128 * 2);
    _Float16* Wp_em2 = (_Float16*)alloc((size_t)GN_L * 128 * 64 * 2);
    _Float16* Wp_nm1 = (_Float16*)alloc((size_t)GN_L * 128 * 128 * 2);
    _Float16* Wp_nm2 = (_Float16*)alloc((size_t)GN_L * 128 * 64 * 2);
    _Float16* Wp_dec = (_Float16*)alloc(64 * 64 * 2);
    _Float16* Wp_out1 = (_Float16*)alloc(128 * 64 * 2);

    _Float16* x16    = (_Float16*)alloc((size_t)GN_N * 128 * 2);
    _Float16* henc16 = (_Float16*)alloc((size_t)GN_N * 64 * 2);
    float*    xh     = (float*)   alloc((size_t)GN_N * 64 * 4);
    _Float16* xh16   = (_Float16*)alloc((size_t)GN_N * 64 * 2);
    float*    ef     = (float*)   alloc((size_t)GN_E * 4 * 4);
    _Float16* hee16  = (_Float16*)alloc((size_t)GN_E * 64 * 2);
    float*    ea     = (float*)   alloc((size_t)GN_E * 64 * 4);
    _Float16* ea16   = (_Float16*)alloc((size_t)GN_E * 64 * 2);
    _Float16* h1_16  = (_Float16*)alloc((size_t)GN_E * 128 * 2);
    float*    agg    = (float*)   alloc((size_t)GN_N * 64 * 4);
    float*    cnt    = (float*)   alloc((size_t)GN_N * 4);
    _Float16* h2_16  = (_Float16*)alloc((size_t)GN_N * 128 * 2);
    float*    gsum   = (float*)   alloc((size_t)GN_G * 64 * 4);
    _Float16* gsum16 = (_Float16*)alloc((size_t)GN_G * 64 * 2);
    float*    graph  = (float*)   alloc((size_t)GN_G * 64 * 4);
    _Float16* graph16 = (_Float16*)alloc((size_t)GN_G * 64 * 2);
    _Float16* emb16  = (_Float16*)alloc((size_t)GN_NE * 64 * 2);
    _Float16* dos_in16 = (_Float16*)alloc((size_t)GN_NE * GN_G * 128 * 2);
    float*    dos_h  = (float*)   alloc((size_t)GN_NE * GN_G * 64 * 4);
    (void)ws_size; (void)n_in; (void)in_sizes; (void)out_size;

    const int B = 256;
    auto g1 = [&](long long n) { return dim3((unsigned)gn_cdiv(n, B)); };

    // ---- weight repack (f32 -> f16, WMMA B-fragment layout) ----
    gn_repack_w<<<g1(128 * 64), B, 0, stream>>>(ne_w1, Wp_ne1, GN_NA, 128, 64);
    gn_repack_w<<<g1(64 * 64), B, 0, stream>>>(ne_w2, Wp_ne2, 64, 64, 64);
    gn_repack_w<<<g1(64 * 64), B, 0, stream>>>(ee_w2, Wp_ee2, 64, 64, 64);
    for (int l = 0; l < GN_L; ++l) {
        gn_repack_w<<<g1(192 * 128), B, 0, stream>>>(em_w1 + (size_t)l * 192 * 128,
                                                     Wp_em1 + (size_t)l * 192 * 128, 192, 192, 128);
        gn_repack_w<<<g1(128 * 64), B, 0, stream>>>(em_w2 + (size_t)l * 128 * 64,
                                                    Wp_em2 + (size_t)l * 128 * 64, 128, 128, 64);
        gn_repack_w<<<g1(128 * 128), B, 0, stream>>>(nm_w1 + (size_t)l * 128 * 128,
                                                     Wp_nm1 + (size_t)l * 128 * 128, 128, 128, 128);
        gn_repack_w<<<g1(128 * 64), B, 0, stream>>>(nm_w2 + (size_t)l * 128 * 64,
                                                    Wp_nm2 + (size_t)l * 128 * 64, 128, 128, 64);
    }
    gn_repack_w<<<g1(64 * 64), B, 0, stream>>>(dec_w, Wp_dec, 64, 64, 64);
    gn_repack_w<<<g1(128 * 64), B, 0, stream>>>(out_w1, Wp_out1, 128, 128, 64);
    gn_f32_to_f16<<<g1(GN_NE * 64), B, 0, stream>>>(emb, emb16, (size_t)GN_NE * 64);

    // ---- edge featurization + degree counts ----
    gn_edge_feat<<<g1(GN_E), B, 0, stream>>>(edge_vec, ef);
    hipMemsetAsync(cnt, 0, (size_t)GN_N * 4, stream);
    gn_count<<<g1(GN_E), B, 0, stream>>>(colp, cnt);

    // ---- node encoder: xh = prelu(x@W1+b1)@W2+b2 (dual f32/f16 store) ----
    gn_pad_x<<<g1((long long)GN_N * 128), B, 0, stream>>>(x, x16);
    const int ntilesN = GN_N / 16;      // 3125
    const int ntilesE = GN_E / 16;      // 25000
    gn_gemm_bias_prelu_f16<4, 4><<<gn_cdiv(ntilesN, 8), 256, 0, stream>>>(
        x16, Wp_ne1, ne_b1, ne_p1, henc16, ntilesN);
    gn_gemm_bias_dual<4, 2><<<gn_cdiv(ntilesN, 8), 256, 0, stream>>>(
        henc16, Wp_ne2, ne_b2, xh, xh16, ntilesN);

    // ---- edge encoder: ea = prelu(ef@W1+b1)@W2+b2 (dual f32/f16 store) ----
    gn_ee1<<<g1((long long)GN_E * 64), B, 0, stream>>>(ef, ee_w1, ee_b1, ee_p1, hee16);
    gn_gemm_bias_dual<4, 2><<<gn_cdiv(ntilesE, 8), 256, 0, stream>>>(
        hee16, Wp_ee2, ee_b2, ea, ea16, ntilesE);

    // ---- message-passing layers ----
    for (int l = 0; l < GN_L; ++l) {
        gn_edge_gemm1_ln_prelu<<<gn_cdiv(ntilesE, 8), 256, 0, stream>>>(
            xh16, ea16, rowp, colp, Wp_em1 + (size_t)l * 192 * 128,
            em_b1 + (size_t)l * 128, em_g + (size_t)l * 128, em_be + (size_t)l * 128,
            em_p + l, h1_16, ntilesE);
        hipMemsetAsync(agg, 0, (size_t)GN_N * 64 * 4, stream);
        gn_edge_gemm2_fused<<<gn_cdiv(ntilesE, 8), 256, 0, stream>>>(
            h1_16, Wp_em2 + (size_t)l * 128 * 64, em_b2 + (size_t)l * 64,
            colp, agg, ea, ea16, ntilesE);
        gn_node_gemm1_ln_prelu<<<gn_cdiv(ntilesN, 8), 256, 0, stream>>>(
            xh16, agg, cnt, Wp_nm1 + (size_t)l * 128 * 128,
            nm_b1 + (size_t)l * 128, nm_g + (size_t)l * 128, nm_be + (size_t)l * 128,
            nm_p + l, h2_16, ntilesN);
        gn_node_gemm2_fused<<<gn_cdiv(ntilesN, 8), 256, 0, stream>>>(
            h2_16, Wp_nm2 + (size_t)l * 128 * 64, nm_b2 + (size_t)l * 64,
            xh, xh16, ntilesN);
    }

    // ---- readout ----
    hipMemsetAsync(gsum, 0, (size_t)GN_G * 64 * 4, stream);
    gn_scatter_batch<<<g1((long long)GN_N * 64), B, 0, stream>>>(xh, batch, gsum);
    gn_f32_to_f16<<<g1(GN_G * 64), B, 0, stream>>>(gsum, gsum16, (size_t)GN_G * 64);
    gn_gemm_bias_f32<4, 2><<<gn_cdiv(GN_G / 16, 8), 256, 0, stream>>>(
        gsum16, Wp_dec, dec_b, graph, GN_G / 16);
    gn_f32_to_f16<<<g1(GN_G * 64), B, 0, stream>>>(graph, graph16, (size_t)GN_G * 64);

    gn_build_dos_in<<<g1((long long)GN_NE * GN_G * 64), B, 0, stream>>>(emb16, graph16, dos_in16);
    const int ntilesD = (GN_NE * GN_G) / 16;   // 816
    gn_gemm_bias_leaky_f32<4, 4><<<gn_cdiv(ntilesD, 8), 256, 0, stream>>>(
        dos_in16, Wp_out1, out_b1, dos_h, ntilesD);
    gn_final<<<g1(GN_NE * GN_G), B, 0, stream>>>(dos_h, out_w2, out_b2, out);
}
